// GraphVS_40346922778948
// MI455X (gfx1250) — matrix operand order repaired and checked by reading, not compiled, not run on hardware
//
#include <hip/hip_runtime.h>
#include <hip/hip_bf16.h>
#include <math.h>

// ---------------- problem constants (from reference) ----------------
#define HD    128       // hidden dim H
#define NPTS  65536     // N0
#define NCL   4096      // NCLU
#define NBAT  16        // B
#define PTSC  16        // points per cluster
#define CPGR  256       // clusters per graph
#define DEGE  32        // ring degree
#define NEDG  (NCL*DEGE)  // 131072 backbone edges

// ---------------- WMMA plumbing (gfx1250, wave32) ----------------
typedef __attribute__((ext_vector_type(16))) __bf16 v16bf;
typedef __attribute__((ext_vector_type(8)))  float  v8f;

// native f32 -> bf16 (RNE fptrunc -> v_cvt_*bf16_f32 on gfx1250)
__device__ __forceinline__ __bf16 f2bf(float f) { return (__bf16)f; }

#define WMMA_BF16(AF, BF, ACC) \
  __builtin_amdgcn_wmma_f32_16x16x32_bf16(false, (AF), false, (BF), (short)0, (ACC), false, false)

// Pack 8 consecutive floats (two float4) into A-fragment group g.
// ISA 7.12.2 A 16x32 bf16: lane {row=lane&15, half=lane>>4}; VGPR v holds K pair
// (v>>2)*16 + half*8 + (v&3)*2  =>  group g = v>>2 covers 8 consecutive k.
__device__ __forceinline__ void packA8(v16bf& af, int g, const float* __restrict__ p) {
  const float4 p0 = *(const float4*)(p);
  const float4 p1 = *(const float4*)(p + 4);
  af[8*g+0] = f2bf(p0.x); af[8*g+1] = f2bf(p0.y);
  af[8*g+2] = f2bf(p0.z); af[8*g+3] = f2bf(p0.w);
  af[8*g+4] = f2bf(p1.x); af[8*g+5] = f2bf(p1.y);
  af[8*g+6] = f2bf(p1.z); af[8*g+7] = f2bf(p1.w);
}

// ---------------------------------------------------------------------------
// Generic C[M,N] = act(A[M,K] @ W[K,N] + bias)   (row-major f32, bf16 WMMA)
// One wave computes a 16x64 output strip (4 x 16x16 tiles, 4 WMMAs per k-step).
// Requires M%16==0, N%64==0.  K handled as full 32-steps + branchless tail.
// ---------------------------------------------------------------------------
__global__ __launch_bounds__(256) void k_gemm(
    const float* __restrict__ A, const float* __restrict__ W,
    const float* __restrict__ bias, float* __restrict__ C,
    int M, int K, int N, int act)
{
  const int lane = threadIdx.x & 31;
  const int wave = threadIdx.x >> 5;
  const int tilesN = N >> 6;                        // 64-wide col groups
  const int tile = blockIdx.x * 8 + wave;           // wave-uniform
  if (tile >= (M >> 4) * tilesN) return;            // wave-uniform exit
  const int rt = tile / tilesN;
  const int cg = tile - rt * tilesN;
  const int l16 = lane & 15;
  const int half = lane >> 4;
  const int col0 = (cg << 6) + l16;
  const float* arow = A + (size_t)((rt << 4) + l16) * K;
  v8f acc0 = {}, acc1 = {}, acc2 = {}, acc3 = {};
  const int Kmain = K & ~31;
  int kb = 0;
  for (; kb < Kmain; kb += 32) {
    v16bf af, bf0, bf1, bf2, bf3;
    packA8(af, 0, arow + kb + (half << 3));
    packA8(af, 1, arow + kb + 16 + (half << 3));
#pragma unroll
    for (int v = 0; v < 8; ++v) {
      const size_t r0 = (size_t)(kb + ((v >> 2) << 4) + (half << 3) + ((v & 3) << 1)) * N + col0;
      const size_t r1 = r0 + N;
      bf0[2*v] = f2bf(W[r0     ]); bf0[2*v+1] = f2bf(W[r1     ]);
      bf1[2*v] = f2bf(W[r0 + 16]); bf1[2*v+1] = f2bf(W[r1 + 16]);
      bf2[2*v] = f2bf(W[r0 + 32]); bf2[2*v+1] = f2bf(W[r1 + 32]);
      bf3[2*v] = f2bf(W[r0 + 48]); bf3[2*v+1] = f2bf(W[r1 + 48]);
    }
    acc0 = WMMA_BF16(af, bf0, acc0);
    acc1 = WMMA_BF16(af, bf1, acc1);
    acc2 = WMMA_BF16(af, bf2, acc2);
    acc3 = WMMA_BF16(af, bf3, acc3);
  }
  if (kb < K) {   // tail: clamped-address loads + selects, no divergent branches
    v16bf af, bf0, bf1, bf2, bf3;
#pragma unroll
    for (int v = 0; v < 8; ++v) {
      const int ka = kb + ((v >> 2) << 4) + (half << 3) + ((v & 3) << 1);
      float a0 = arow[ka     < K ? ka     : 0]; if (ka     >= K) a0 = 0.f;
      float a1 = arow[ka + 1 < K ? ka + 1 : 0]; if (ka + 1 >= K) a1 = 0.f;
      af[2*v] = f2bf(a0); af[2*v+1] = f2bf(a1);
      const int k0 = kb + ((v >> 2) << 4) + (half << 3) + ((v & 3) << 1);
      const int k1 = k0 + 1;
      const size_t r0 = (size_t)(k0 < K ? k0 : 0) * N + col0;
      const size_t r1 = (size_t)(k1 < K ? k1 : 0) * N + col0;
      const float z0 = (k0 < K) ? 1.f : 0.f, z1 = (k1 < K) ? 1.f : 0.f;
      bf0[2*v] = f2bf(W[r0     ] * z0); bf0[2*v+1] = f2bf(W[r1     ] * z1);
      bf1[2*v] = f2bf(W[r0 + 16] * z0); bf1[2*v+1] = f2bf(W[r1 + 16] * z1);
      bf2[2*v] = f2bf(W[r0 + 32] * z0); bf2[2*v+1] = f2bf(W[r1 + 32] * z1);
      bf3[2*v] = f2bf(W[r0 + 48] * z0); bf3[2*v+1] = f2bf(W[r1 + 48] * z1);
    }
    acc0 = WMMA_BF16(af, bf0, acc0);
    acc1 = WMMA_BF16(af, bf1, acc1);
    acc2 = WMMA_BF16(af, bf2, acc2);
    acc3 = WMMA_BF16(af, bf3, acc3);
  }
#pragma unroll
  for (int v = 0; v < 8; ++v) {
    const size_t row = (size_t)((rt << 4) + (half << 3) + v) * N + col0;
    float y0 = acc0[v] + (bias ? bias[col0     ] : 0.f);
    float y1 = acc1[v] + (bias ? bias[col0 + 16] : 0.f);
    float y2 = acc2[v] + (bias ? bias[col0 + 32] : 0.f);
    float y3 = acc3[v] + (bias ? bias[col0 + 48] : 0.f);
    if (act == 1) { y0 = fmaxf(y0, 0.f); y1 = fmaxf(y1, 0.f); y2 = fmaxf(y2, 0.f); y3 = fmaxf(y3, 0.f); }
    C[row] = y0; C[row + 16] = y1; C[row + 32] = y2; C[row + 48] = y3;
  }
}

// ---------------------------------------------------------------------------
// Fused edge-message GEMM: for edge e with j=src[e], i=dst[e]:
//   feat(k) = X[i][k] (k<Hc) | X[j][k-Hc]-X[i][k-Hc] (k<2Hc) | pos[j]-pos[i] (k<2Hc+3)
// C[e,:] = feat @ W[2Hc+3, N] + bias.  Hc%32==0 => region picked by wave-uniform kb.
// One wave computes 16 edges x 64 cols.  Requires N%64==0.
// ---------------------------------------------------------------------------
__global__ __launch_bounds__(256) void k_msg_gemm(
    const float* __restrict__ X, const float* __restrict__ pos,
    const int* __restrict__ src, const int* __restrict__ dst,
    const float* __restrict__ W, const float* __restrict__ bias,
    float* __restrict__ C, int E, int Hc, int N)
{
  const int lane = threadIdx.x & 31;
  const int wave = threadIdx.x >> 5;
  const int tilesN = N >> 6;
  const int tile = blockIdx.x * 8 + wave;
  if (tile >= (E >> 4) * tilesN) return;
  const int rt = tile / tilesN;
  const int cg = tile - rt * tilesN;
  const int l16 = lane & 15;
  const int half = lane >> 4;
  const int col0 = (cg << 6) + l16;
  const int e = (rt << 4) + l16;
  const int i = dst[e], j = src[e];
  const float* xi = X + (size_t)i * Hc;
  const float* xj = X + (size_t)j * Hc;
  const float dp0 = pos[j*3+0] - pos[i*3+0];
  const float dp1 = pos[j*3+1] - pos[i*3+1];
  const float dp2 = pos[j*3+2] - pos[i*3+2];
  const int K = 2 * Hc + 3;
  v8f acc0 = {}, acc1 = {}, acc2 = {}, acc3 = {};

  for (int kb = 0; kb < 2 * Hc; kb += 32) {     // full steps, region uniform per kb
    v16bf af, bf0, bf1, bf2, bf3;
    if (kb + 32 <= Hc) {                        // region 0: x_i
      packA8(af, 0, xi + kb + (half << 3));
      packA8(af, 1, xi + kb + 16 + (half << 3));
    } else {                                    // region 1: x_j - x_i
      const int kk = kb - Hc;
#pragma unroll
      for (int g = 0; g < 2; ++g) {
        const int off = kk + (g << 4) + (half << 3);
        const float4 a0 = *(const float4*)(xi + off);
        const float4 a1 = *(const float4*)(xi + off + 4);
        const float4 b0 = *(const float4*)(xj + off);
        const float4 b1 = *(const float4*)(xj + off + 4);
        af[8*g+0] = f2bf(b0.x - a0.x); af[8*g+1] = f2bf(b0.y - a0.y);
        af[8*g+2] = f2bf(b0.z - a0.z); af[8*g+3] = f2bf(b0.w - a0.w);
        af[8*g+4] = f2bf(b1.x - a1.x); af[8*g+5] = f2bf(b1.y - a1.y);
        af[8*g+6] = f2bf(b1.z - a1.z); af[8*g+7] = f2bf(b1.w - a1.w);
      }
    }
#pragma unroll
    for (int v = 0; v < 8; ++v) {
      const size_t r0 = (size_t)(kb + ((v >> 2) << 4) + (half << 3) + ((v & 3) << 1)) * N + col0;
      const size_t r1 = r0 + N;
      bf0[2*v] = f2bf(W[r0     ]); bf0[2*v+1] = f2bf(W[r1     ]);
      bf1[2*v] = f2bf(W[r0 + 16]); bf1[2*v+1] = f2bf(W[r1 + 16]);
      bf2[2*v] = f2bf(W[r0 + 32]); bf2[2*v+1] = f2bf(W[r1 + 32]);
      bf3[2*v] = f2bf(W[r0 + 48]); bf3[2*v+1] = f2bf(W[r1 + 48]);
    }
    acc0 = WMMA_BF16(af, bf0, acc0);
    acc1 = WMMA_BF16(af, bf1, acc1);
    acc2 = WMMA_BF16(af, bf2, acc2);
    acc3 = WMMA_BF16(af, bf3, acc3);
  }
  {   // final step kb = 2Hc: only k-2Hc in {0,1,2} valid (pos diffs, in registers)
    const int kb = 2 * Hc;
    v16bf af, bf0, bf1, bf2, bf3;
#pragma unroll
    for (int v = 0; v < 8; ++v) {
#pragma unroll
      for (int q = 0; q < 2; ++q) {
        const int kk = ((v >> 2) << 4) + (half << 3) + ((v & 3) << 1) + q;  // k - 2Hc
        float f = (kk == 0) ? dp0 : (kk == 1) ? dp1 : (kk == 2) ? dp2 : 0.f;
        af[2*v + q] = f2bf(f);
      }
      const int k0 = kb + ((v >> 2) << 4) + (half << 3) + ((v & 3) << 1);
      const int k1 = k0 + 1;
      const size_t r0 = (size_t)(k0 < K ? k0 : 0) * N + col0;
      const size_t r1 = (size_t)(k1 < K ? k1 : 0) * N + col0;
      const float z0 = (k0 < K) ? 1.f : 0.f, z1 = (k1 < K) ? 1.f : 0.f;
      bf0[2*v] = f2bf(W[r0     ] * z0); bf0[2*v+1] = f2bf(W[r1     ] * z1);
      bf1[2*v] = f2bf(W[r0 + 16] * z0); bf1[2*v+1] = f2bf(W[r1 + 16] * z1);
      bf2[2*v] = f2bf(W[r0 + 32] * z0); bf2[2*v+1] = f2bf(W[r1 + 32] * z1);
      bf3[2*v] = f2bf(W[r0 + 48] * z0); bf3[2*v+1] = f2bf(W[r1 + 48] * z1);
    }
    acc0 = WMMA_BF16(af, bf0, acc0);
    acc1 = WMMA_BF16(af, bf1, acc1);
    acc2 = WMMA_BF16(af, bf2, acc2);
    acc3 = WMMA_BF16(af, bf3, acc3);
  }
  const float b0v = bias ? bias[col0     ] : 0.f;
  const float b1v = bias ? bias[col0 + 16] : 0.f;
  const float b2v = bias ? bias[col0 + 32] : 0.f;
  const float b3v = bias ? bias[col0 + 48] : 0.f;
#pragma unroll
  for (int v = 0; v < 8; ++v) {
    const size_t row = (size_t)((rt << 4) + (half << 3) + v) * N + col0;
    C[row] = acc0[v] + b0v; C[row + 16] = acc1[v] + b1v;
    C[row + 32] = acc2[v] + b2v; C[row + 48] = acc3[v] + b3v;
  }
}

// ---------------- elementwise / reduction helpers ----------------
__global__ void k_gather_rows(const float* __restrict__ src, const int* __restrict__ idx,
                              float* __restrict__ dst, int n, int cols) {
  int t = blockIdx.x * blockDim.x + threadIdx.x;
  if (t >= n * cols) return;
  int r = t / cols, c = t - r * cols;
  dst[t] = src[(size_t)idx[r] * cols + c];
}

// pos-diff for PTC edges (e01: j=e, i=e>>4). mode0: src=[pos_cur,pos_tar]; mode1: [pos_tar,pos_tar]
__global__ void k_build_pd(const float* __restrict__ pc, const float* __restrict__ pt,
                           const float* __restrict__ pctr, float* __restrict__ pd,
                           int n0, int mode) {
  int t = blockIdx.x * blockDim.x + threadIdx.x;
  if (t >= n0 * 6) return;
  int e = t / 6, c = t - e * 6;
  int i = e >> 4;
  float s;
  if (c < 3) s = (mode ? pt : pc)[(size_t)e * 3 + c];
  else       s = pt[(size_t)e * 3 + (c - 3)];
  pd[t] = pctr[(size_t)i * 3 + (c % 3)] - s;
}

// a_in[e,ch] = q[e>>4, ch] - k[e,ch] + delta[e,ch]
__global__ void k_ain(const float* __restrict__ q, const float* __restrict__ kk,
                      const float* __restrict__ dl, float* __restrict__ out, int n0) {
  int t = blockIdx.x * blockDim.x + threadIdx.x;
  if (t >= n0 * HD) return;
  int e = t / HD, ch = t - e * HD;
  out[t] = q[(size_t)(e >> 4) * HD + ch] - kk[t] + dl[t];
}

// per-channel softmax over 16 consecutive edges of each cluster, in place
__global__ void k_softmax16(float* __restrict__ a, int ncl) {
  int t = blockIdx.x * blockDim.x + threadIdx.x;
  if (t >= ncl * HD) return;
  int c = t / HD, ch = t - c * HD;
  size_t base = (size_t)c * 16 * HD + ch;
  float m = -3.0e38f;
  for (int r = 0; r < 16; ++r) m = fmaxf(m, a[base + (size_t)r * HD]);
  float s = 0.f;
  for (int r = 0; r < 16; ++r) s += __expf(a[base + (size_t)r * HD] - m);
  float inv = 1.f / s;
  for (int r = 0; r < 16; ++r) {
    size_t p = base + (size_t)r * HD;
    a[p] = __expf(a[p] - m) * inv;
  }
}

// out[c,ch] = relu( sum_r a[e,ch]*(v[e,ch]+delta[e,ch]) ), e = c*16+r
__global__ void k_ptc_reduce(const float* __restrict__ a, const float* __restrict__ v,
                             const float* __restrict__ dl, float* __restrict__ out, int ncl) {
  int t = blockIdx.x * blockDim.x + threadIdx.x;
  if (t >= ncl * HD) return;
  int c = t / HD, ch = t - c * HD;
  size_t base = (size_t)c * 16 * HD + ch;
  float s = 0.f;
  for (int r = 0; r < 16; ++r) {
    size_t p = base + (size_t)r * HD;
    s += a[p] * (v[p] + dl[p]);
  }
  out[t] = fmaxf(s, 0.f);
}

// mode0: out=[a,b]; mode1: out=[a, b-a]
__global__ void k_concat2(const float* __restrict__ a, const float* __restrict__ b,
                          float* __restrict__ out, int n, int c, int mode) {
  int t = blockIdx.x * blockDim.x + threadIdx.x;
  if (t >= n * 2 * c) return;
  int r = t / (2 * c), ch = t - r * 2 * c;
  float y;
  if (ch < c) y = a[(size_t)r * c + ch];
  else {
    float bv = b[(size_t)r * c + (ch - c)];
    y = mode ? (bv - a[(size_t)r * c + (ch - c)]) : bv;
  }
  out[t] = y;
}

__global__ void k_mask_mul(float* __restrict__ x, const unsigned char* __restrict__ mask, int n) {
  int t = blockIdx.x * blockDim.x + threadIdx.x;
  if (t >= n * HD) return;
  if (!mask[t / HD]) x[t] = 0.f;
}

// segment-max over ring edges e = d*NCL + c; finite->value else 0; act: 0 none, 2 sigmoid, 3 tanh
__global__ void k_segmax_ring(const float* __restrict__ msg, float* __restrict__ out,
                              int nch, int act) {
  int t = blockIdx.x * blockDim.x + threadIdx.x;
  if (t >= NCL * nch) return;
  int c = t / nch, ch = t - c * nch;
  float m = -3.0e38f;
  for (int d = 0; d < DEGE; ++d)
    m = fmaxf(m, msg[((size_t)d * NCL + c) * nch + ch]);
  if (!(m > -3.0e38f && m < 3.0e38f) || m != m) m = 0.f;
  if (act == 2) m = 1.f / (1.f + __expf(-m));
  else if (act == 3) m = tanhf(m);
  out[t] = m;
}

// graph-norm stats per (graph, channel): stats[0..] = mean, stats[NG*HD..] = var of (x-mean*ms)
__global__ void k_gn_stats(const float* __restrict__ x, const float* __restrict__ ms,
                           float* __restrict__ stats) {
  int t = blockIdx.x * blockDim.x + threadIdx.x;
  if (t >= NBAT * HD) return;
  int g = t / HD, ch = t - g * HD;
  size_t base = (size_t)g * CPGR * HD + ch;
  float sum = 0.f;
  for (int c = 0; c < CPGR; ++c) sum += x[base + (size_t)c * HD];
  float mean = sum / (float)CPGR;
  float mm = mean * ms[ch];
  float var = 0.f;
  for (int c = 0; c < CPGR; ++c) {
    float o = x[base + (size_t)c * HD] - mm;
    var += o * o;
  }
  stats[t] = mean;
  stats[NBAT * HD + t] = var / (float)CPGR;
}

__global__ void k_gn_apply(const float* __restrict__ x, const float* __restrict__ stats,
                           const float* __restrict__ w, const float* __restrict__ b,
                           const float* __restrict__ ms, float* __restrict__ out) {
  int t = blockIdx.x * blockDim.x + threadIdx.x;
  if (t >= NCL * HD) return;
  int c = t / HD, ch = t - c * HD;
  int g = c / CPGR;
  float mean = stats[g * HD + ch];
  float var  = stats[NBAT * HD + g * HD + ch];
  float o = x[t] - mean * ms[ch];
  float y = w[ch] * o * rsqrtf(var + 1e-5f) + b[ch];
  out[t] = fmaxf(y, 0.f);   // relu after graph_norm (per_conv)
}

__global__ void k_add_relu(const float* __restrict__ a, const float* __restrict__ b,
                           float* __restrict__ out, int n) {
  int t = blockIdx.x * blockDim.x + threadIdx.x;
  if (t < n) out[t] = fmaxf(a[t] + b[t], 0.f);
}

// comb2 = [xb0, hidden * r]  where r = gates[:, :HD]
__global__ void k_comb2(const float* __restrict__ xb0, const float* __restrict__ h,
                        const float* __restrict__ gates, float* __restrict__ out) {
  int t = blockIdx.x * blockDim.x + threadIdx.x;
  if (t >= NCL * 2 * HD) return;
  int c = t / (2 * HD), ch = t - c * 2 * HD;
  if (ch < HD) out[t] = xb0[(size_t)c * HD + ch];
  else {
    int k = ch - HD;
    out[t] = h[(size_t)c * HD + k] * gates[(size_t)c * 2 * HD + k];
  }
}

// hnew = (1-z)*h + z*ht,  z = gates[:, HD:]
__global__ void k_gru(const float* __restrict__ h, const float* __restrict__ gates,
                      const float* __restrict__ ht, float* __restrict__ hnew) {
  int t = blockIdx.x * blockDim.x + threadIdx.x;
  if (t >= NCL * HD) return;
  int c = t / HD, ch = t - c * HD;
  float z = gates[(size_t)c * 2 * HD + HD + ch];
  hnew[t] = (1.f - z) * h[t] + z * ht[t];
}

__global__ void k_logits(const float* __restrict__ x, const float* __restrict__ w,
                         const float* __restrict__ b, const unsigned char* __restrict__ mask,
                         float* __restrict__ lg) {
  int c = blockIdx.x * blockDim.x + threadIdx.x;
  if (c >= NCL) return;
  float s = b[0];
  for (int k = 0; k < HD; ++k) s += x[(size_t)c * HD + k] * w[k];
  lg[c] = mask[c] ? s : -1e30f;
}

__global__ void k_softmax_graph(const float* __restrict__ lg, const unsigned char* __restrict__ mask,
                                float* __restrict__ gate) {
  int g = threadIdx.x;
  if (g >= NBAT) return;
  const float* l = lg + (size_t)g * CPGR;
  float m = -3.0e38f;
  for (int c = 0; c < CPGR; ++c) m = fmaxf(m, l[c]);
  float s = 0.f;
  for (int c = 0; c < CPGR; ++c) s += __expf(l[c] - m);
  float inv = 1.f / s;
  for (int c = 0; c < CPGR; ++c)
    gate[g * CPGR + c] = mask[g * CPGR + c] ? __expf(l[c] - m) * inv : 0.f;
}

__global__ void k_scene(const float* __restrict__ gate, const float* __restrict__ x,
                        float* __restrict__ scene) {
  int t = blockIdx.x * blockDim.x + threadIdx.x;
  if (t >= NBAT * HD) return;
  int g = t / HD, ch = t - g * HD;
  float s = 0.f;
  for (int c = 0; c < CPGR; ++c) {
    int row = g * CPGR + c;
    s += gate[row] * x[(size_t)row * HD + ch];
  }
  scene[t] = s;
}

// tiny scalar GEMM for N not multiple of 16 (final heads)
__global__ void k_small(const float* __restrict__ A, const float* __restrict__ W,
                        const float* __restrict__ b, float* __restrict__ C,
                        int M, int K, int N) {
  int t = blockIdx.x * blockDim.x + threadIdx.x;
  if (t >= M * N) return;
  int m = t / N, n = t - m * N;
  float s = b ? b[n] : 0.f;
  for (int k = 0; k < K; ++k) s += A[(size_t)m * K + k] * W[(size_t)k * N + n];
  C[t] = s;
}

// ---------------------------------------------------------------------------
extern "C" void kernel_launch(void* const* d_in, const int* in_sizes, int n_in,
                              void* d_out, int out_size, void* d_ws, size_t ws_size,
                              hipStream_t stream) {
  (void)in_sizes; (void)n_in; (void)out_size; (void)ws_size;
  // inputs in setup_inputs() dict order, params dict expanded in insertion order
  const float* x_cur   = (const float*)d_in[0];
  const float* x_tar   = (const float*)d_in[1];
  const float* pos_cur = (const float*)d_in[2];
  const float* pos_tar = (const float*)d_in[3];
  const float* hidden  = (const float*)d_in[4];
  const float* in_w = (const float*)d_in[5],  *in_b = (const float*)d_in[6];
  const float* ptc_lin_w = (const float*)d_in[7], *ptc_src_w = (const float*)d_in[8],
             * ptc_dst_w = (const float*)d_in[9];
  const float* pos_w1 = (const float*)d_in[10], *pos_b1 = (const float*)d_in[11];
  const float* pos_w2 = (const float*)d_in[12], *pos_b2 = (const float*)d_in[13];
  const float* att_w1 = (const float*)d_in[14], *att_b1 = (const float*)d_in[15];
  const float* att_w2 = (const float*)d_in[16], *att_b2 = (const float*)d_in[17];
  const float* out_w = (const float*)d_in[18], *out_b = (const float*)d_in[19];
  const float* c0_msg_w = (const float*)d_in[20], *c0_msg_b = (const float*)d_in[21];
  const float* c0_gn_w = (const float*)d_in[22], *c0_gn_b = (const float*)d_in[23],
             * c0_gn_ms = (const float*)d_in[24];
  const float* c0_fc_w = (const float*)d_in[25], *c0_fc_b = (const float*)d_in[26];
  const float* c1_msg_w = (const float*)d_in[27], *c1_msg_b = (const float*)d_in[28];
  const float* c1_gn_w = (const float*)d_in[29], *c1_gn_b = (const float*)d_in[30],
             * c1_gn_ms = (const float*)d_in[31];
  const float* c1_fc_w = (const float*)d_in[32], *c1_fc_b = (const float*)d_in[33];
  const float* g_gate_w = (const float*)d_in[34], *g_gate_b = (const float*)d_in[35];
  const float* g_cand_w = (const float*)d_in[36], *g_cand_b = (const float*)d_in[37];
  const float* gate_w = (const float*)d_in[38], *gate_b = (const float*)d_in[39];
  const float* vec_w1 = (const float*)d_in[40], *vec_b1 = (const float*)d_in[41];
  const float* vec_w2 = (const float*)d_in[42], *vec_b2 = (const float*)d_in[43];
  const float* nrm_w1 = (const float*)d_in[44], *nrm_b1 = (const float*)d_in[45];
  const float* nrm_w2 = (const float*)d_in[46], *nrm_b2 = (const float*)d_in[47];
  const int* ei_cur  = (const int*)d_in[49];
  const int* ei_tar  = (const int*)d_in[50];
  const int* centers = (const int*)d_in[51];
  const unsigned char* cmask = (const unsigned char*)d_in[52];
  float* out = (float*)d_out;
  float* HNEW = out + NBAT * 6 + NBAT * 1;   // hidden output region (offset 112)

  // ---- workspace layout (floats) ----
  float* ws = (float*)d_ws;
  size_t o = 0;
  auto alloc = [&](size_t n) { size_t r = o; o += (n + 1023) & ~(size_t)1023; return ws + r; };
  float* XC    = alloc((size_t)NPTS * HD);
  float* XT    = alloc((size_t)NPTS * HD);
  float* XDST  = alloc((size_t)NCL * HD);
  float* Q     = alloc((size_t)NCL * HD);
  float* PCTR  = alloc((size_t)NCL * 3);
  float* XCC_C = alloc((size_t)NCL * HD);
  float* XCC_T = alloc((size_t)NCL * HD);
  float* XCAT  = alloc((size_t)NCL * 2 * HD);
  float* XCLU  = alloc((size_t)NCL * HD);
  float* XB0   = alloc((size_t)NCL * HD);
  float* COMB  = alloc((size_t)NCL * 2 * HD);
  float* GATES = alloc((size_t)NCL * 2 * HD);
  float* HT    = alloc((size_t)NCL * HD);
  float* AGG   = alloc((size_t)NCL * HD);
  float* GNS   = alloc((size_t)NBAT * HD * 2);
  float* TMP   = alloc((size_t)NCL * HD);
  float* FCO   = alloc((size_t)NCL * HD);
  float* XOUT  = alloc((size_t)NCL * HD);
  float* SCN   = alloc((size_t)NBAT * HD);
  float* LOG   = alloc((size_t)NCL);
  float* GTE   = alloc((size_t)NCL);
  float* V1    = alloc((size_t)NBAT * HD);
  float* N1    = alloc((size_t)NBAT * HD);
  float* PD    = alloc((size_t)NPTS * 6);
  float* EDGE  = alloc((size_t)5 * NPTS * HD);   // stage-1 scratch, reused as MSG later
  float* KB = EDGE;
  float* VB = EDGE + (size_t)1 * NPTS * HD;
  float* DL = EDGE + (size_t)2 * NPTS * HD;
  float* T1 = EDGE + (size_t)3 * NPTS * HD;
  float* T2 = EDGE + (size_t)4 * NPTS * HD;
  float* MSG = EDGE;                              // up to 131072 x 256 = 32M floats

  auto GEMM = [&](const float* A, const float* W, const float* b, float* C,
                  int M, int K, int N, int act) {
    int tiles = (M / 16) * (N / 64);
    k_gemm<<<dim3((tiles + 7) / 8), dim3(256), 0, stream>>>(A, W, b, C, M, K, N, act);
  };
  auto MSGG = [&](const float* X, const int* ei, const float* W, const float* b,
                  float* C, int Hc, int N) {
    int tiles = (NEDG / 16) * (N / 64);
    k_msg_gemm<<<dim3((tiles + 7) / 8), dim3(256), 0, stream>>>(
        X, PCTR, ei, ei + NEDG, W, b, C, NEDG, Hc, N);
  };
  auto EW = [](size_t n) { return dim3((unsigned)((n + 255) / 256)); };

  // ---- encoder ----
  GEMM(x_cur, in_w, in_b, XC, NPTS, 32, HD, 1);
  GEMM(x_tar, in_w, in_b, XT, NPTS, 32, HD, 1);

  // ---- PTC shared: centers gather + q ----
  k_gather_rows<<<EW((size_t)NCL * HD), 256, 0, stream>>>(XT, centers, XDST, NCL, HD);
  k_gather_rows<<<EW((size_t)NCL * 3), 256, 0, stream>>>(pos_tar, centers, PCTR, NCL, 3);
  GEMM(XDST, ptc_dst_w, nullptr, Q, NCL, HD, HD, 0);

  // ---- PTC: pass 0 = current, pass 1 = target ----
  for (int pass = 0; pass < 2; ++pass) {
    const float* xs = pass ? XT : XC;
    k_build_pd<<<EW((size_t)NPTS * 6), 256, 0, stream>>>(pos_cur, pos_tar, PCTR, PD, NPTS, pass);
    GEMM(xs, ptc_src_w, nullptr, KB, NPTS, HD, HD, 0);
    GEMM(xs, ptc_lin_w, nullptr, VB, NPTS, HD, HD, 0);
    GEMM(PD, pos_w1, pos_b1, T1, NPTS, 6, HD, 1);
    GEMM(T1, pos_w2, pos_b2, DL, NPTS, HD, HD, 0);
    k_ain<<<EW((size_t)NPTS * HD), 256, 0, stream>>>(Q, KB, DL, T2, NPTS);
    GEMM(T2, att_w1, att_b1, T1, NPTS, HD, HD, 1);
    GEMM(T1, att_w2, att_b2, T2, NPTS, HD, HD, 0);
    k_softmax16<<<EW((size_t)NCL * HD), 256, 0, stream>>>(T2, NCL);
    k_ptc_reduce<<<EW((size_t)NCL * HD), 256, 0, stream>>>(T2, VB, DL, pass ? XCC_T : XCC_C, NCL);
  }

  // ---- combine + out linear + mask ----
  k_concat2<<<EW((size_t)NCL * 2 * HD), 256, 0, stream>>>(XCC_C, XCC_T, XCAT, NCL, HD, 1);
  GEMM(XCAT, out_w, out_b, XCLU, NCL, 2 * HD, HD, 1);
  k_mask_mul<<<EW((size_t)NCL * HD), 256, 0, stream>>>(XCLU, cmask, NCL);

  // ---- per_conv c0 ----
  MSGG(XCLU, ei_cur, c0_msg_w, c0_msg_b, MSG, HD, HD);
  k_segmax_ring<<<EW((size_t)NCL * HD), 256, 0, stream>>>(MSG, AGG, HD, 0);
  k_gn_stats<<<EW((size_t)NBAT * HD), 256, 0, stream>>>(AGG, c0_gn_ms, GNS);
  k_gn_apply<<<EW((size_t)NCL * HD), 256, 0, stream>>>(AGG, GNS, c0_gn_w, c0_gn_b, c0_gn_ms, TMP);
  GEMM(TMP, c0_fc_w, c0_fc_b, FCO, NCL, HD, HD, 0);
  k_add_relu<<<EW((size_t)NCL * HD), 256, 0, stream>>>(FCO, XCLU, XB0, NCL * HD);

  // ---- GRU ----
  k_concat2<<<EW((size_t)NCL * 2 * HD), 256, 0, stream>>>(XB0, hidden, COMB, NCL, HD, 0);
  MSGG(COMB, ei_tar, g_gate_w, g_gate_b, MSG, 2 * HD, 2 * HD);
  k_segmax_ring<<<EW((size_t)NCL * 2 * HD), 256, 0, stream>>>(MSG, GATES, 2 * HD, 2);
  k_comb2<<<EW((size_t)NCL * 2 * HD), 256, 0, stream>>>(XB0, hidden, GATES, COMB);
  MSGG(COMB, ei_cur, g_cand_w, g_cand_b, MSG, 2 * HD, HD);
  k_segmax_ring<<<EW((size_t)NCL * HD), 256, 0, stream>>>(MSG, HT, HD, 3);
  k_gru<<<EW((size_t)NCL * HD), 256, 0, stream>>>(hidden, GATES, HT, HNEW);

  // ---- per_conv c1 on new hidden ----
  MSGG(HNEW, ei_cur, c1_msg_w, c1_msg_b, MSG, HD, HD);
  k_segmax_ring<<<EW((size_t)NCL * HD), 256, 0, stream>>>(MSG, AGG, HD, 0);
  k_gn_stats<<<EW((size_t)NBAT * HD), 256, 0, stream>>>(AGG, c1_gn_ms, GNS);
  k_gn_apply<<<EW((size_t)NCL * HD), 256, 0, stream>>>(AGG, GNS, c1_gn_w, c1_gn_b, c1_gn_ms, TMP);
  GEMM(TMP, c1_fc_w, c1_fc_b, FCO, NCL, HD, HD, 0);
  k_add_relu<<<EW((size_t)NCL * HD), 256, 0, stream>>>(FCO, HNEW, XOUT, NCL * HD);

  // ---- decoder ----
  k_logits<<<EW((size_t)NCL), 256, 0, stream>>>(XOUT, gate_w, gate_b, cmask, LOG);
  k_softmax_graph<<<dim3(1), dim3(32), 0, stream>>>(LOG, cmask, GTE);
  k_scene<<<EW((size_t)NBAT * HD), 256, 0, stream>>>(GTE, XOUT, SCN);
  GEMM(SCN, vec_w1, vec_b1, V1, NBAT, HD, HD, 1);
  k_small<<<dim3(1), dim3(256), 0, stream>>>(V1, vec_w2, vec_b2, out, NBAT, HD, 6);
  GEMM(SCN, nrm_w1, nrm_b1, N1, NBAT, HD, HD, 1);
  k_small<<<dim3(1), dim3(256), 0, stream>>>(N1, nrm_w2, nrm_b2, out + NBAT * 6, NBAT, HD, 1);
}
